// Q_DimeNetPP_5952824672739
// MI455X (gfx1250) — matrix operand order repaired and checked by reading, not compiled, not run on hardware
//
#include <hip/hip_runtime.h>
#include <hip/hip_bf16.h>
#include <cstdint>
#include <cstddef>

// ---------------- problem constants (match reference) ----------------
#define N_NODES 20000
#define E_EDGES 300000
#define T_TRIP  400000
#define H_DIM   256
#define IE_DIM  128
#define OE_DIM  256
#define OC_DIM  256
#define NGRAPHS 64
#define PI_F    3.14159265358979323846f

typedef __attribute__((ext_vector_type(16))) _Float16 v16h;
typedef __attribute__((ext_vector_type(8)))  _Float16 v8h;
typedef __attribute__((ext_vector_type(8)))  float    v8f;
typedef __attribute__((ext_vector_type(4)))  float    v4f;

// ---------------- device helpers ----------------
__device__ __forceinline__ float siluf(float v)  { return v / (1.f + __expf(-v)); }
__device__ __forceinline__ float lreluf(float v) { return v > 0.f ? v : 0.01f * v; }
__device__ __forceinline__ float actf(float v, int act) {
    if (act == 1) return siluf(v);
    if (act == 2) return lreluf(v);
    return v;
}

// ---------------- utility kernels ----------------
__global__ void fill_kernel(float* __restrict__ p, float v, int n) {
    int i = blockIdx.x * 256 + threadIdx.x;
    if (i < n) p[i] = v;
}

// pack f32 W[K][Nc] (row major) -> f16 Wp[Nc][K] (col of W contiguous)
__global__ void pack_w_kernel(const float* __restrict__ W, _Float16* __restrict__ Wp,
                              int K, int Nc) {
    int i = blockIdx.x * 256 + threadIdx.x;
    if (i >= K * Nc) return;
    int n = i / K, k = i - n * K;
    Wp[i] = (_Float16)W[(size_t)k * Nc + n];
}

// small-K dense GEMM (VALU): C = act(A[M,K] @ W[K,Nc] + bias)
__global__ void gemm_small_kernel(const float* __restrict__ A, const float* __restrict__ W,
                                  const float* __restrict__ bias, float* __restrict__ C,
                                  int M, int K, int Nc, int act) {
    long long i = (long long)blockIdx.x * 256 + threadIdx.x;
    if (i >= (long long)M * Nc) return;
    int row = (int)(i / Nc);
    int col = (int)(i - (long long)row * Nc);
    float v = bias ? bias[col] : 0.f;
    const float* a = A + (size_t)row * K;
    for (int k = 0; k < K; ++k) v += a[k] * W[(size_t)k * Nc + col];
    C[i] = actf(v, act);
}

// geometry: gaussian rbf9, dvec = pos[tar]-pos[src], bessel rbf with envelope
__global__ void geom_kernel(const float* __restrict__ pos, const int* __restrict__ src,
                            const int* __restrict__ tar, float* __restrict__ rbf9,
                            float* __restrict__ dvec, float* __restrict__ rbf) {
    int e = blockIdx.x * 256 + threadIdx.x;
    if (e >= E_EDGES) return;
    int s = src[e], t = tar[e];
    float dx = pos[t * 3 + 0] - pos[s * 3 + 0];
    float dy = pos[t * 3 + 1] - pos[s * 3 + 1];
    float dz = pos[t * 3 + 2] - pos[s * 3 + 2];
    float d  = sqrtf(dx * dx + dy * dy + dz * dz);
    dvec[e * 3 + 0] = dx; dvec[e * 3 + 1] = dy; dvec[e * 3 + 2] = dz;
    const float inv_sig = 9.f / 6.f;
#pragma unroll
    for (int r = 0; r < 9; ++r) {
        float u = (d - 0.75f * (float)r) * inv_sig;
        rbf9[e * 9 + r] = __expf(-u * u);
    }
    float du = fmaxf(d * 0.2f, 1e-2f);
    float env = 0.f;
    if (du < 1.f) {
        float d2 = du * du, d4 = d2 * d2, d5 = d4 * du;
        env = 1.f / du - 28.f * d5 + 48.f * d5 * du - 21.f * d5 * d2;   // p=6 envelope
    }
#pragma unroll
    for (int r = 0; r < 9; ++r) rbf[e * 9 + r] = env * __sinf((float)(r + 1) * PI_F * du);
}

// fused conv message: agg[tar] += x1[src] * silu(rbf9 @ base_W + base_b)
__global__ void conv_scatter_kernel(const float* __restrict__ rbf9, const float* __restrict__ baseW,
                                    const float* __restrict__ baseb, const float* __restrict__ x1,
                                    const int* __restrict__ src, const int* __restrict__ tar,
                                    float* __restrict__ agg) {
    long long i = (long long)blockIdx.x * 256 + threadIdx.x;
    if (i >= (long long)E_EDGES * H_DIM) return;
    int e = (int)(i >> 8), h = (int)(i & (H_DIM - 1));
    float g = baseb[h];
#pragma unroll
    for (int r = 0; r < 9; ++r) g += rbf9[e * 9 + r] * baseW[r * H_DIM + h];
    g = siluf(g);
    atomicAdd(&agg[(size_t)tar[e] * H_DIM + h], x1[(size_t)src[e] * H_DIM + h] * g);
}

__global__ void bn_prep_kernel(const float* __restrict__ g, const float* __restrict__ b,
                               const float* __restrict__ m, const float* __restrict__ v,
                               float* __restrict__ scl, float* __restrict__ shf) {
    int h = threadIdx.x;
    if (h >= H_DIM) return;
    float s = g[h] * rsqrtf(v[h] + 1e-5f);
    scl[h] = s;
    shf[h] = b[h] - m[h] * s;
}

__global__ void cnt_accum_kernel(const int* __restrict__ iji, float* __restrict__ cnt) {
    int t = blockIdx.x * 256 + threadIdx.x;
    if (t < T_TRIP) atomicAdd(&cnt[iji[t]], 1.f);
}
__global__ void cnt_clamp_kernel(float* __restrict__ cnt) {
    int e = blockIdx.x * 256 + threadIdx.x;
    if (e < E_EDGES) cnt[e] = fmaxf(cnt[e], 1.f);
}

// e = silu(P0[tar] + P1[src] + eacc + lin_b)
__global__ void e_combine_kernel(const float* __restrict__ P0, const float* __restrict__ P1,
                                 const float* __restrict__ eacc, const float* __restrict__ linb,
                                 const int* __restrict__ src, const int* __restrict__ tar,
                                 float* __restrict__ eo) {
    long long i = (long long)blockIdx.x * 256 + threadIdx.x;
    if (i >= (long long)E_EDGES * H_DIM) return;
    int e = (int)(i >> 8), h = (int)(i & (H_DIM - 1));
    float v = P0[(size_t)tar[e] * H_DIM + h] + P1[(size_t)src[e] * H_DIM + h] + eacc[i] + linb[h];
    eo[i] = siluf(v);
}

// tacc[tar] += (rbf @ obW) * e
__global__ void ob_scatter_kernel(const float* __restrict__ rbf, const float* __restrict__ obW,
                                  const float* __restrict__ e_, const int* __restrict__ tar,
                                  float* __restrict__ tacc) {
    long long i = (long long)blockIdx.x * 256 + threadIdx.x;
    if (i >= (long long)E_EDGES * H_DIM) return;
    int e = (int)(i >> 8), h = (int)(i & (H_DIM - 1));
    float g = 0.f;
#pragma unroll
    for (int r = 0; r < 9; ++r) g += rbf[e * 9 + r] * obW[r * H_DIM + h];
    atomicAdd(&tacc[(size_t)tar[e] * H_DIM + h], g * e_[i]);
}

// per-triplet: tmp8[t][q] = sum_{s,r} cos(s*angle)*rad[r]*sbf1[(s*9+r)][q]
__global__ void tmp8_kernel(const float* __restrict__ dvec, const float* __restrict__ rbf,
                            const int* __restrict__ ikj, const int* __restrict__ iji,
                            const float* __restrict__ sbf1, float* __restrict__ tmp8) {
    int t = blockIdx.x * 256 + threadIdx.x;
    if (t >= T_TRIP) return;
    int ej = iji[t], ek = ikj[t];
    float ax = dvec[ej * 3], ay = dvec[ej * 3 + 1], az = dvec[ej * 3 + 2];
    float bx = dvec[ek * 3], by = dvec[ek * 3 + 1], bz = dvec[ek * 3 + 2];
    float cx = ay * bz - az * by, cy = az * bx - ax * bz, cz = ax * by - ay * bx;
    float cn = sqrtf(cx * cx + cy * cy + cz * cz);
    float dt = ax * bx + ay * by + az * bz;
    float ang = atan2f(cn, dt);
    float rad[9];
#pragma unroll
    for (int r = 0; r < 9; ++r) rad[r] = rbf[ek * 9 + r];
    float cs[7];
#pragma unroll
    for (int s = 0; s < 7; ++s) cs[s] = __cosf((float)s * ang);
#pragma unroll
    for (int q = 0; q < 8; ++q) {
        float acc = 0.f;
        for (int s = 0; s < 7; ++s) {
            float tt = 0.f;
#pragma unroll
            for (int r = 0; r < 9; ++r) tt += rad[r] * sbf1[((s * 9 + r) << 3) + q];
            acc += cs[s] * tt;
        }
        tmp8[(size_t)t * 8 + q] = acc;
    }
}

// magg[iji] += xdown[ikj] * (tmp8 @ sbf2)
__global__ void triplet_scatter_kernel(const float* __restrict__ tmp8, const float* __restrict__ sbf2,
                                       const float* __restrict__ xdown, const int* __restrict__ ikj,
                                       const int* __restrict__ iji, float* __restrict__ magg) {
    long long i = (long long)blockIdx.x * 256 + threadIdx.x;
    if (i >= (long long)T_TRIP * IE_DIM) return;
    int t = (int)(i >> 7), ie = (int)(i & (IE_DIM - 1));
    float v = 0.f;
#pragma unroll
    for (int q = 0; q < 8; ++q) v += tmp8[(size_t)t * 8 + q] * sbf2[q * IE_DIM + ie];
    atomicAdd(&magg[(size_t)iji[t] * IE_DIM + ie], xdown[(size_t)ikj[t] * IE_DIM + ie] * v);
}

__global__ void seg_batch_kernel(const float* __restrict__ x1, const int* __restrict__ batch,
                                 float* __restrict__ outG) {
    long long i = (long long)blockIdx.x * 256 + threadIdx.x;
    if (i >= (long long)N_NODES * H_DIM) return;
    int n = (int)(i >> 8), h = (int)(i & (H_DIM - 1));
    atomicAdd(&outG[(size_t)batch[n] * H_DIM + h], x1[i]);
}

// gated attention over {x1,x2} : one block of 128 threads per node (LDS reduction)
__global__ __launch_bounds__(128) void attn_kernel(
    const float* __restrict__ Va, const float* __restrict__ Ua,
    const float* __restrict__ Vb, const float* __restrict__ Ub,
    const float* __restrict__ wv, const float* __restrict__ x1,
    const float* __restrict__ x2, float* __restrict__ pvec) {
    int n = blockIdx.x, j = threadIdx.x;
    __shared__ float red0[128], red1[128];
    __shared__ float aw0, aw1;
    float wj = wv[j];
    size_t o = (size_t)n * 128 + j;
    red0[j] = tanhf(Va[o]) * (1.f / (1.f + __expf(-Ua[o]))) * wj;
    red1[j] = tanhf(Vb[o]) * (1.f / (1.f + __expf(-Ub[o]))) * wj;
    __syncthreads();
    for (int off = 64; off > 0; off >>= 1) {
        if (j < off) { red0[j] += red0[j + off]; red1[j] += red1[j + off]; }
        __syncthreads();
    }
    if (j == 0) {
        float m = fmaxf(red0[0], red1[0]);
        float e0 = __expf(red0[0] - m), e1 = __expf(red1[0] - m);
        float inv = 1.f / (e0 + e1);
        aw0 = e0 * inv; aw1 = e1 * inv;
    }
    __syncthreads();
    for (int h = j; h < H_DIM; h += 128) {
        size_t p = (size_t)n * H_DIM + h;
        pvec[p] = aw0 * x1[p] + aw1 * x2[p];
    }
}

// ---------------- WMMA GEMM ----------------
// C[M,Nc] = epilogue( (A (+A2)) * (1/rowDiv) @ Wf16[Nc][K] )
// epilogue: +bias -> act -> (*gate8) -> (*scl+shf) -> (+resid)
// 256 threads = 8 wave32; block tile 128x64; wave tile 32x32 (2x2 subtiles of 16x16);
// LDS A tile double-buffered -> one barrier per K-step; 4 v_wmma per wave per K-step.
#define LDA 40                           // halves per LDS row (80B, 16B-aligned groups)
__global__ __launch_bounds__(256) void gemm_wmma_kernel(
    const float* __restrict__ A, const float* __restrict__ A2,
    const float* __restrict__ rowDiv,
    const _Float16* __restrict__ Wp, const float* __restrict__ bias,
    const float* __restrict__ gA8, const float* __restrict__ gW8,
    const float* __restrict__ resid,
    const float* __restrict__ scl, const float* __restrict__ shf,
    float* __restrict__ C, int M, int K, int Nc, int act) {
    __shared__ _Float16 As[2 * 128 * LDA];           // double-buffered 128x32 f16 tile
    const int tid  = threadIdx.x;
    const int lane = tid & 31, wave = tid >> 5;
    const int m16  = lane & 15, hi = lane >> 4;
    const int waveM = wave & 3, waveN = wave >> 2;   // waveM: 32-row group, waveN: 32-col group
    const int row0 = blockIdx.x * 128;
    const int col0 = blockIdx.y * 64;

    v8f acc00 = {}, acc01 = {}, acc10 = {}, acc11 = {};

    // staging: 256 threads x 2 tasks cover 128 rows x 4 k-groups of 8 floats
    const int srow = tid >> 2;                       // 0..63
    const int akg  = tid & 3;
    const int growA = row0 + srow;
    const int growB = growA + 64;
    const bool rokA = growA < M;
    const bool rokB = growB < M;
    float rsA = 1.f, rsB = 1.f;
    if (rowDiv) {
        if (rokA) rsA = 1.f / rowDiv[growA];
        if (rokB) rsB = 1.f / rowDiv[growB];
    }

    auto stage = [&](int buf, int k0) {
        _Float16* dst = As + buf * (128 * LDA);
#pragma unroll
        for (int half = 0; half < 2; ++half) {
            const int  grow = half ? growB : growA;
            const bool rok  = half ? rokB  : rokA;
            const float rs  = half ? rsB   : rsA;
            v4f x0 = {}, x1v = {};
            if (rok) {
                const v4f* p = (const v4f*)(A + (size_t)grow * K + k0 + akg * 8);
                x0 = p[0]; x1v = p[1];
                if (A2) {
                    const v4f* q = (const v4f*)(A2 + (size_t)grow * K + k0 + akg * 8);
                    v4f y0 = q[0], y1 = q[1];
                    x0 += y0; x1v += y1;
                }
                if (k0 + 32 < K) __builtin_prefetch(A + (size_t)grow * K + k0 + 32, 0, 1);
            }
            v8h hh;
#pragma unroll
            for (int i = 0; i < 4; ++i) {
                hh[i]     = (_Float16)(x0[i]  * rs);
                hh[i + 4] = (_Float16)(x1v[i] * rs);
            }
            *(v8h*)(&dst[(srow + half * 64) * LDA + akg * 8]) = hh;
        }
    };

    const int nsteps = K >> 5;
    stage(0, 0);
    __syncthreads();

    for (int it = 0; it < nsteps; ++it) {
        const int cur = it & 1;
        const int k0  = it << 5;
        if (it + 1 < nsteps) stage(cur ^ 1, k0 + 32);

        union HF { v16h v; v8h h[2]; };
        HF af0, af1, bf0, bf1;
        // A fragments (ISA layout: halves 0..7 <- K +8*hi, 8..15 <- K +16+8*hi)
        const _Float16* ap0 = &As[cur * (128 * LDA) + (waveM * 32 + m16) * LDA];
        const _Float16* ap1 = ap0 + 16 * LDA;
        af0.h[0] = *(const v8h*)(ap0 + 8 * hi);
        af0.h[1] = *(const v8h*)(ap0 + 16 + 8 * hi);
        af1.h[0] = *(const v8h*)(ap1 + 8 * hi);
        af1.h[1] = *(const v8h*)(ap1 + 16 + 8 * hi);
        // B fragments (ISA layout: halves 0..7 <- K +8*hi, 8..15 <- K +16+8*hi)
        const int colA = col0 + waveN * 32 + m16;
        const _Float16* wp0 = Wp + (size_t)colA * K + k0 + 8 * hi;
        const _Float16* wp1 = Wp + (size_t)(colA + 16) * K + k0 + 8 * hi;
        bf0.h[0] = *(const v8h*)(wp0);  bf0.h[1] = *(const v8h*)(wp0 + 16);
        bf1.h[0] = *(const v8h*)(wp1);  bf1.h[1] = *(const v8h*)(wp1 + 16);

        acc00 = __builtin_amdgcn_wmma_f32_16x16x32_f16(false, af0.v, false, bf0.v,
                                                       (short)0, acc00, false, false);
        acc01 = __builtin_amdgcn_wmma_f32_16x16x32_f16(false, af0.v, false, bf1.v,
                                                       (short)0, acc01, false, false);
        acc10 = __builtin_amdgcn_wmma_f32_16x16x32_f16(false, af1.v, false, bf0.v,
                                                       (short)0, acc10, false, false);
        acc11 = __builtin_amdgcn_wmma_f32_16x16x32_f16(false, af1.v, false, bf1.v,
                                                       (short)0, acc11, false, false);
        __syncthreads();
    }

#pragma unroll
    for (int mi = 0; mi < 2; ++mi) {
#pragma unroll
        for (int r = 0; r < 8; ++r) {
            const int row = row0 + waveM * 32 + mi * 16 + r + 8 * hi;
            if (row >= M) continue;
#pragma unroll
            for (int ni = 0; ni < 2; ++ni) {
                const int col = col0 + waveN * 32 + ni * 16 + m16;
                float v = mi ? (ni ? acc11[r] : acc10[r]) : (ni ? acc01[r] : acc00[r]);
                if (bias) v += bias[col];
                v = actf(v, act);
                if (gA8) {
                    float g = 0.f;
#pragma unroll
                    for (int q = 0; q < 8; ++q) g += gA8[(size_t)row * 8 + q] * gW8[q * Nc + col];
                    v *= g;
                }
                if (scl) v = v * scl[col] + shf[col];
                if (resid) v += resid[(size_t)row * Nc + col];
                C[(size_t)row * Nc + col] = v;
            }
        }
    }
}

// ---------------- host orchestration ----------------
static inline void launch_gemm(hipStream_t st, const float* A, const float* A2,
                               const float* rowDiv, const _Float16* Wp, const float* bias,
                               const float* gA8, const float* gW8, const float* resid,
                               const float* scl, const float* shf, float* C,
                               int M, int K, int Nc, int act) {
    dim3 g((M + 127) / 128, Nc / 64);
    gemm_wmma_kernel<<<g, 256, 0, st>>>(A, A2, rowDiv, Wp, bias, gA8, gW8, resid,
                                        scl, shf, C, M, K, Nc, act);
}

extern "C" void kernel_launch(void* const* d_in, const int* in_sizes, int n_in,
                              void* d_out, int out_size, void* d_ws, size_t ws_size,
                              hipStream_t stream) {
    (void)in_sizes; (void)n_in; (void)out_size; (void)ws_size;
    auto F = [&](int i) { return (const float*)d_in[i]; };
    const int* eidx  = (const int*)d_in[3];
    const int* src   = eidx;             // edge_index[0]
    const int* tar   = eidx + E_EDGES;   // edge_index[1]
    const int* batch = (const int*)d_in[4];
    const int* ikj   = (const int*)d_in[5];
    const int* iji   = (const int*)d_in[6];

    // ---- workspace carve (deterministic) ----
    char* wsp = (char*)d_ws;
    auto allocf = [&](size_t n) -> float* {
        float* p = (float*)wsp; wsp += ((n * 4 + 255) / 256) * 256; return p;
    };
    auto alloch = [&](size_t n) -> _Float16* {
        _Float16* p = (_Float16*)wsp; wsp += ((n * 2 + 255) / 256) * 256; return p;
    };
    auto pack = [&](const float* W, int K, int Nc) -> _Float16* {
        _Float16* p = alloch((size_t)K * Nc);
        int n = K * Nc;
        pack_w_kernel<<<(n + 255) / 256, 256, 0, stream>>>(W, p, K, Nc);
        return p;
    };
    auto fill = [&](float* p, size_t n, float v) {
        fill_kernel<<<(unsigned)((n + 255) / 256), 256, 0, stream>>>(p, v, (int)n);
    };

    // ---- pack all WMMA weights to f16 [Nc][K] ----
    _Float16* covP[3];
    for (int c = 0; c < 3; ++c) covP[c] = pack(F(9 + 8 * c + 2), H_DIM, H_DIM);
    const float* linW = F(37);  // emb2 lin_W [768,256]
    _Float16* embP0 = pack(linW,                 H_DIM, H_DIM);
    _Float16* embP1 = pack(linW + 256 * H_DIM,   H_DIM, H_DIM);
    _Float16* embP2 = pack(linW + 512 * H_DIM,   H_DIM, H_DIM);
    _Float16 *kjP[3], *jiP[3], *dnP[3], *upP[3], *lnP[3];
    for (int b = 0; b < 3; ++b) {
        int base = 39 + 12 * b;
        kjP[b] = pack(F(base + 4), H_DIM, H_DIM);
        jiP[b] = pack(F(base + 6), H_DIM, H_DIM);
        dnP[b] = pack(F(base + 8), H_DIM, IE_DIM);
        upP[b] = pack(F(base + 9), IE_DIM, H_DIM);
        lnP[b] = pack(F(base + 10), H_DIM, H_DIM);
    }
    _Float16 *obUpP[4], *obL0P[4], *obL1P[4], *obL2P[4], *obLinP[4];
    for (int o = 0; o < 4; ++o) {
        int base = 75 + 9 * o;
        obUpP[o]  = pack(F(base + 1), H_DIM, OE_DIM);
        obL0P[o]  = pack(F(base + 2), OE_DIM, OE_DIM);
        obL1P[o]  = pack(F(base + 4), OE_DIM, OE_DIM);
        obL2P[o]  = pack(F(base + 6), OE_DIM, OE_DIM);
        obLinP[o] = pack(F(base + 8), OE_DIM, OC_DIM);
    }
    _Float16* boP[3];
    for (int c = 0; c < 3; ++c) boP[c] = pack(F(111 + 2 * c), H_DIM, OC_DIM);
    _Float16* attnVP = pack(F(117), H_DIM, 128);
    _Float16* attnUP = pack(F(118), H_DIM, 128);

    // ---- activation buffers ----
    float* X1A  = allocf((size_t)N_NODES * H_DIM);
    float* X1B  = allocf((size_t)N_NODES * H_DIM);
    float* AGG  = allocf((size_t)N_NODES * H_DIM);
    float* SCL  = allocf(H_DIM);
    float* SHF  = allocf(H_DIM);
    float* RBF9 = allocf((size_t)E_EDGES * 9);
    float* DVEC = allocf((size_t)E_EDGES * 3);
    float* RBF  = allocf((size_t)E_EDGES * 9);
    float* CNTC = allocf(E_EDGES);
    float* HN   = allocf((size_t)N_NODES * H_DIM);
    float* P0   = allocf((size_t)N_NODES * H_DIM);
    float* P1   = allocf((size_t)N_NODES * H_DIM);
    float* E0   = allocf((size_t)E_EDGES * H_DIM);
    float* E1   = allocf((size_t)E_EDGES * H_DIM);
    float* XJI  = allocf((size_t)E_EDGES * H_DIM);   // doubles as he
    float* XKJ  = allocf((size_t)E_EDGES * H_DIM);   // doubles as eacc
    float* XDN  = allocf((size_t)E_EDGES * IE_DIM);
    float* MAGG = allocf((size_t)E_EDGES * IE_DIM);
    float* R8   = allocf((size_t)E_EDGES * 8);
    float* TMP8 = allocf((size_t)T_TRIP * 8);
    float* TACC = allocf((size_t)N_NODES * H_DIM);
    float* TN0  = allocf((size_t)N_NODES * OE_DIM);
    float* TN1  = allocf((size_t)N_NODES * OE_DIM);
    float* X2A  = allocf((size_t)N_NODES * OC_DIM);
    float* X2B  = allocf((size_t)N_NODES * OC_DIM);
    float* VA   = allocf((size_t)N_NODES * 128);
    float* UA   = allocf((size_t)N_NODES * 128);
    float* VB   = allocf((size_t)N_NODES * 128);
    float* UB   = allocf((size_t)N_NODES * 128);

    const long long NH = (long long)N_NODES * H_DIM;
    const long long EH = (long long)E_EDGES * H_DIM;

    // ---- geometry + radial bases ----
    geom_kernel<<<(E_EDGES + 255) / 256, 256, 0, stream>>>(F(1), src, tar, RBF9, DVEC, RBF);

    // ---- x1 = silu(x @ lin_node) ----
    gemm_small_kernel<<<(unsigned)((NH + 255) / 256), 256, 0, stream>>>(
        F(0), F(7), F(8), X1A, N_NODES, 44, H_DIM, 1);

    // ---- 3x InteractionLayer convs ----
    float *x1c = X1A, *x1n = X1B;
    for (int c = 0; c < 3; ++c) {
        int base = 9 + 8 * c;
        fill(AGG, (size_t)NH, 0.f);
        conv_scatter_kernel<<<(unsigned)((EH + 255) / 256), 256, 0, stream>>>(
            RBF9, F(base), F(base + 1), x1c, src, tar, AGG);
        bn_prep_kernel<<<1, 256, 0, stream>>>(F(base + 4), F(base + 5), F(base + 6),
                                              F(base + 7), SCL, SHF);
        launch_gemm(stream, x1c, AGG, nullptr, covP[c], F(base + 3), nullptr, nullptr,
                    nullptr, SCL, SHF, x1n, N_NODES, H_DIM, H_DIM, 2);
        float* t = x1c; x1c = x1n; x1n = t;
    }
    const float* x1f = x1c;   // final x1

    // ---- triplet counts ----
    fill(CNTC, E_EDGES, 0.f);
    cnt_accum_kernel<<<(T_TRIP + 255) / 256, 256, 0, stream>>>(iji, CNTC);
    cnt_clamp_kernel<<<(E_EDGES + 255) / 256, 256, 0, stream>>>(CNTC);

    // ---- embedding block (K=768 GEMM decomposed) ----
    gemm_small_kernel<<<(unsigned)((NH + 255) / 256), 256, 0, stream>>>(
        F(0), F(33), F(34), HN, N_NODES, 44, H_DIM, 1);
    gemm_small_kernel<<<(unsigned)((EH + 255) / 256), 256, 0, stream>>>(
        F(2), F(35), F(36), XJI, E_EDGES, 12, H_DIM, 1);           // he
    launch_gemm(stream, HN, nullptr, nullptr, embP0, nullptr, nullptr, nullptr,
                nullptr, nullptr, nullptr, P0, N_NODES, H_DIM, H_DIM, 0);
    launch_gemm(stream, HN, nullptr, nullptr, embP1, nullptr, nullptr, nullptr,
                nullptr, nullptr, nullptr, P1, N_NODES, H_DIM, H_DIM, 0);
    launch_gemm(stream, XJI, nullptr, nullptr, embP2, nullptr, nullptr, nullptr,
                nullptr, nullptr, nullptr, XKJ, E_EDGES, H_DIM, H_DIM, 0);  // eacc
    e_combine_kernel<<<(unsigned)((EH + 255) / 256), 256, 0, stream>>>(
        P0, P1, XKJ, F(38), src, tar, E0);
    float *ec = E0, *eo = E1;

    // ---- output block helper ----
    auto run_ob = [&](int o, const float* ein, float* dst) {
        int base = 75 + 9 * o;
        fill(TACC, (size_t)NH, 0.f);
        ob_scatter_kernel<<<(unsigned)((EH + 255) / 256), 256, 0, stream>>>(
            RBF, F(base), ein, tar, TACC);
        launch_gemm(stream, TACC, nullptr, nullptr, obUpP[o], nullptr, nullptr, nullptr,
                    nullptr, nullptr, nullptr, TN0, N_NODES, H_DIM, OE_DIM, 0);
        launch_gemm(stream, TN0, nullptr, nullptr, obL0P[o], F(base + 3), nullptr, nullptr,
                    nullptr, nullptr, nullptr, TN1, N_NODES, OE_DIM, OE_DIM, 1);
        launch_gemm(stream, TN1, nullptr, nullptr, obL1P[o], F(base + 5), nullptr, nullptr,
                    nullptr, nullptr, nullptr, TN0, N_NODES, OE_DIM, OE_DIM, 1);
        launch_gemm(stream, TN0, nullptr, nullptr, obL2P[o], F(base + 7), nullptr, nullptr,
                    nullptr, nullptr, nullptr, TN1, N_NODES, OE_DIM, OE_DIM, 1);
        launch_gemm(stream, TN1, nullptr, nullptr, obLinP[o], nullptr, nullptr, nullptr,
                    nullptr, nullptr, nullptr, dst, N_NODES, OE_DIM, OC_DIM, 0);
    };

    run_ob(0, ec, X2A);
    float *x2c = X2A, *x2n = X2B;

    // ---- 3x DimeNet interaction blocks ----
    for (int b = 0; b < 3; ++b) {
        int base = 39 + 12 * b;
        launch_gemm(stream, ec, nullptr, nullptr, jiP[b], F(base + 7), nullptr, nullptr,
                    nullptr, nullptr, nullptr, XJI, E_EDGES, H_DIM, H_DIM, 1);
        gemm_small_kernel<<<(unsigned)(((long long)E_EDGES * 8 + 255) / 256), 256, 0, stream>>>(
            RBF, F(base + 0), nullptr, R8, E_EDGES, 9, 8, 0);
        launch_gemm(stream, ec, nullptr, nullptr, kjP[b], F(base + 5), R8, F(base + 1),
                    nullptr, nullptr, nullptr, XKJ, E_EDGES, H_DIM, H_DIM, 1);
        launch_gemm(stream, XKJ, nullptr, nullptr, dnP[b], nullptr, nullptr, nullptr,
                    nullptr, nullptr, nullptr, XDN, E_EDGES, H_DIM, IE_DIM, 1);
        fill(MAGG, (size_t)E_EDGES * IE_DIM, 0.f);
        tmp8_kernel<<<(T_TRIP + 255) / 256, 256, 0, stream>>>(DVEC, RBF, ikj, iji,
                                                              F(base + 2), TMP8);
        triplet_scatter_kernel<<<(unsigned)(((long long)T_TRIP * IE_DIM + 255) / 256), 256, 0,
                                 stream>>>(TMP8, F(base + 3), XDN, ikj, iji, MAGG);
        launch_gemm(stream, MAGG, nullptr, CNTC, upP[b], nullptr, nullptr, nullptr,
                    nullptr, nullptr, nullptr, XKJ, E_EDGES, IE_DIM, H_DIM, 1);
        launch_gemm(stream, XJI, XKJ, nullptr, lnP[b], F(base + 11), nullptr, nullptr,
                    ec, nullptr, nullptr, eo, E_EDGES, H_DIM, H_DIM, 1);
        { float* t = ec; ec = eo; eo = t; }
        run_ob(b + 1, ec, TN0);   // final out-block result lands in TN0
        launch_gemm(stream, x2c, TN0, nullptr, boP[b], F(112 + 2 * b), nullptr, nullptr,
                    nullptr, nullptr, nullptr, x2n, N_NODES, H_DIM, OC_DIM, 2);
        { float* t = x2c; x2c = x2n; x2n = t; }
    }
    const float* x2f = x2c;

    // ---- gated attention + outputs ----
    launch_gemm(stream, x1f, nullptr, nullptr, attnVP, nullptr, nullptr, nullptr,
                nullptr, nullptr, nullptr, VA, N_NODES, H_DIM, 128, 0);
    launch_gemm(stream, x1f, nullptr, nullptr, attnUP, nullptr, nullptr, nullptr,
                nullptr, nullptr, nullptr, UA, N_NODES, H_DIM, 128, 0);
    launch_gemm(stream, x2f, nullptr, nullptr, attnVP, nullptr, nullptr, nullptr,
                nullptr, nullptr, nullptr, VB, N_NODES, H_DIM, 128, 0);
    launch_gemm(stream, x2f, nullptr, nullptr, attnUP, nullptr, nullptr, nullptr,
                nullptr, nullptr, nullptr, UB, N_NODES, H_DIM, 128, 0);

    float* outG = (float*)d_out;                    // [NGRAPHS,H]
    float* pvec = outG + (size_t)NGRAPHS * H_DIM;   // [N,H]
    attn_kernel<<<N_NODES, 128, 0, stream>>>(VA, UA, VB, UB, F(119), x1f, x2f, pvec);

    fill(outG, (size_t)NGRAPHS * H_DIM, 0.f);
    seg_batch_kernel<<<(unsigned)((NH + 255) / 256), 256, 0, stream>>>(x1f, batch, outG);
}
